// MultiHeadAttention_39324720562623
// MI455X (gfx1250) — matrix operand order repaired and checked by reading, not compiled, not run on hardware
//
#include <hip/hip_runtime.h>
#include <hip/hip_bf16.h>

#define NB  4
#define TT  2048
#define DD  512
#define HH  8
#define HDD 64
#define MROWS (NB * TT)   // 8192

typedef __attribute__((ext_vector_type(16))) __bf16        v16bf;
typedef __attribute__((ext_vector_type(8)))  float         v8f;
typedef __attribute__((ext_vector_type(8)))  unsigned int  v8u;

union Frag {
    v8u            u;
    v16bf          b;
    unsigned short h[16];
};

__device__ __forceinline__ unsigned short f2bf(float f) {
    unsigned int u = __float_as_uint(f);
    unsigned int r = u + 0x7FFFu + ((u >> 16) & 1u);   // round-to-nearest-even
    return (unsigned short)(r >> 16);
}
__device__ __forceinline__ float bf2f(unsigned short h) {
    return __uint_as_float(((unsigned int)h) << 16);
}

// ---- CDNA5 async global->LDS copy (ASYNCcnt-tracked, bypasses VGPRs) ----
// generic LDS pointer: addr[31:0] is the LDS byte offset (ISA 10.2 aperture map)
__device__ __forceinline__ void async_copy_b128(void* lds_dst, const void* gbase,
                                                unsigned byte_off) {
    unsigned lds_off = (unsigned)(size_t)lds_dst;
    asm volatile("global_load_async_to_lds_b128 %0, %1, %2"
                 :: "v"(lds_off), "v"(byte_off),
                    "s"((unsigned long long)gbase)
                 : "memory");
}
__device__ __forceinline__ void async_wait0() {
    asm volatile("s_wait_asynccnt 0x0" ::: "memory");
}

// ---------------------------------------------------------------- casts ----
__global__ void cast_f32_bf16_kernel(const float* __restrict__ in,
                                     unsigned short* __restrict__ out, int n) {
    int i = blockIdx.x * blockDim.x + threadIdx.x;
    if (i < n) out[i] = f2bf(in[i]);
}

// ----------------------------------------------------------------- GEMM ----
// Computes C = A @ W^T  (A: [M x 512] bf16, W row-major [n][k] bf16)
// Internally evaluates C^T = W x A^T so each lane owns one OUTPUT ROW and the
// epilogue is packed b128 stores.  Double-buffered async-DMA pipeline:
// stage(i+1) overlaps compute(i); one barrier per K-step.
// MODE 0: bf16 ; 1: bf16(+Res) ; 2: fp32.
#define GM 128
#define GN 64
#define GK 64
#define KSTEPS (DD / GK)   // 8
#define PADG 72   // halfwords per 64-wide LDS row (144 B: 16B-aligned, conflict-free)

template <int MODE>
__global__ __launch_bounds__(256) void gemm_bf16_kernel(
    const unsigned short* __restrict__ A,
    const unsigned short* __restrict__ Wb,    // [DD x DD] row-major bf16
    const unsigned short* __restrict__ Res,
    unsigned short* __restrict__ Obf,
    float* __restrict__ Of)
{
    __shared__ unsigned short sA[2][GM][PADG];
    __shared__ unsigned short sW[2][GN][PADG];

    const int t    = threadIdx.x;
    const int wave = t >> 5;
    const int lane = t & 31;
    const int m0   = blockIdx.y * GM;
    const int n0   = blockIdx.x * GN;

    auto stage = [&](int k0, int b) {
        {   // A tile: 128 rows x 64 k, 4x b128 per thread
            int row = t >> 1;
            int ch  = (t & 1) * 32;
            const unsigned short* gb = A + (size_t)m0 * DD + k0;
            unsigned off = (unsigned)((row * DD + ch) * 2);
            async_copy_b128(&sA[b][row][ch],      gb, off);
            async_copy_b128(&sA[b][row][ch + 8],  gb, off + 16);
            async_copy_b128(&sA[b][row][ch + 16], gb, off + 32);
            async_copy_b128(&sA[b][row][ch + 24], gb, off + 48);
        }
        {   // W tile: 64 rows (n) x 64 k, 2x b128 per thread
            int row = t >> 2;
            int kc  = (t & 3) * 16;
            const unsigned short* gb = Wb + (size_t)n0 * DD + k0;
            unsigned off = (unsigned)((row * DD + kc) * 2);
            async_copy_b128(&sW[b][row][kc],     gb, off);
            async_copy_b128(&sW[b][row][kc + 8], gb, off + 16);
        }
    };

    // C^T tiles: row n = mt*16 + r + (lane>>4)*8 ; col m = wave*16 + (lane&15)
    v8f acc[4];
#pragma unroll
    for (int i = 0; i < 4; i++)
#pragma unroll
        for (int r = 0; r < 8; r++) acc[i][r] = 0.f;

    stage(0, 0);
    for (int kk = 0; kk < KSTEPS; kk++) {
        const int b = kk & 1;
        async_wait0();       // stage(kk) DMA complete (this wave)
        __syncthreads();     // all waves: tile kk in LDS, compute(kk-1) done
        if (kk + 1 < KSTEPS) stage((kk + 1) * GK, b ^ 1);   // overlaps compute

#pragma unroll
        for (int s = 0; s < 2; s++) {
            Frag fx;   // B operand: wave's 16 A-rows as output columns
            {
                int m   = wave * 16 + (lane & 15);
                int kb2 = s * 32 + (lane >> 4) * 16;
#pragma unroll
                for (int j = 0; j < 8; j++)
                    fx.u[j] = *(const unsigned int*)&sA[b][m][kb2 + 2 * j];
            }
#pragma unroll
            for (int mt = 0; mt < 4; mt++) {
                Frag fw;   // A operand: W rows (n as WMMA-M)
                int n  = mt * 16 + (lane & 15);
                int kb = s * 32 + (lane >> 4) * 16;
#pragma unroll
                for (int j = 0; j < 4; j++) {
                    fw.u[j]     = *(const unsigned int*)&sW[b][n][kb + 2 * j];
                    fw.u[4 + j] = *(const unsigned int*)&sW[b][n][kb + 8 + 2 * j];
                }
                acc[mt] = __builtin_amdgcn_wmma_f32_16x16x32_bf16(
                    false, fw.b, false, fx.b, (short)0, acc[mt], false, false);
            }
        }
    }

    {   // epilogue: each lane owns one output row; cols packed as b128
        int m  = m0 + wave * 16 + (lane & 15);
        int cb = (lane >> 4) * 8;
#pragma unroll
        for (int mt = 0; mt < 4; mt++) {
            size_t base = (size_t)m * DD + n0 + mt * 16 + cb;
            if (MODE == 2) {
                union { uint4 v[2]; float f[8]; } pk;
#pragma unroll
                for (int r = 0; r < 8; r++) pk.f[r] = acc[mt][r];
                *(uint4*)(Of + base)     = pk.v[0];
                *(uint4*)(Of + base + 4) = pk.v[1];
            } else if (MODE == 1) {
                union { uint4 v; unsigned short h[8]; } rs, pk;
                rs.v = *(const uint4*)(Res + base);
#pragma unroll
                for (int r = 0; r < 8; r++)
                    pk.h[r] = f2bf(acc[mt][r] + bf2f(rs.h[r]));
                *(uint4*)(Obf + base) = pk.v;
            } else {
                union { uint4 v; unsigned short h[8]; } pk;
#pragma unroll
                for (int r = 0; r < 8; r++) pk.h[r] = f2bf(acc[mt][r]);
                *(uint4*)(Obf + base) = pk.v;
            }
        }
    }
}

// ------------------------------------------------------- flash attention ----
// S^T = K_tile x Q  puts the QUERY in the lane -> softmax is within-lane plus
// a single xor-16 shuffle; O^T = V^T x P^T keeps everything shuffle-free.
// log2(e) folded into the Q scale so softmax uses raw v_exp_f32 (exp2).
// Double-buffered async pipeline: stage(i+1) overlaps compute(i).
#define BQ   128
#define BK   64
#define NIT  (TT / BK)   // 32
#define PADR 72   // halfwords per 64-wide LDS row (144 B: 16B-aligned, conflict-free)

__global__ __launch_bounds__(256) void flash_attn_kernel(
    const unsigned short* __restrict__ Qb,
    const unsigned short* __restrict__ Kb,
    const unsigned short* __restrict__ Vb,
    unsigned short* __restrict__ Ob)
{
    __shared__ unsigned short sQ[BQ][PADR];       // reused as P^T ([query][key])
    __shared__ unsigned short sK[2][BK][PADR];
    __shared__ unsigned short sVt[2][HDD][PADR];  // V transposed: [hd][key]

    const int t    = threadIdx.x;
    const int wave = t >> 5;
    const int lane = t & 31;
    const int qblk = blockIdx.x;
    const int nb   = blockIdx.y;
    const int h    = blockIdx.z;

    const size_t rowbase = (size_t)nb * TT;
    const int    colbase = h * HDD;
    const int    qown    = wave * 16 + (lane & 15);   // lane's query row

    auto stageK = [&](int i, int b) {
        int row = t >> 2;
        int ch  = (t & 3) * 16;
        const unsigned short* gb = Kb + (rowbase + (size_t)i * BK) * DD + colbase;
        unsigned off = (unsigned)((row * DD + ch) * 2);
        async_copy_b128(&sK[b][row][ch],     gb, off);
        async_copy_b128(&sK[b][row][ch + 8], gb, off + 16);
    };

    union V16 { uint4 v[2]; unsigned short e[16]; };
    V16 vreg;
    auto loadV = [&](int i) {
        int row = t >> 2;
        int ch  = (t & 3) * 16;
        const uint4* g = (const uint4*)(Vb + (rowbase + (size_t)i * BK + row) * DD + colbase + ch);
        vreg.v[0] = g[0];
        vreg.v[1] = g[1];
    };
    auto storeV = [&](int b) {   // transpose on store: sVt[hd][key]
        int row = t >> 2;
        int ch  = (t & 3) * 16;
#pragma unroll
        for (int j = 0; j < 16; j++) sVt[b][ch + j][row] = vreg.e[j];
    };

    {   // Q tile: 128 rows x 64 cols, async DMA, 4x b128 per thread
        int row = t >> 1;
        int ch  = (t & 1) * 32;
        const unsigned short* gb = Qb + (rowbase + (size_t)qblk * BQ) * DD + colbase;
        unsigned off = (unsigned)((row * DD + ch) * 2);
        async_copy_b128(&sQ[row][ch],      gb, off);
        async_copy_b128(&sQ[row][ch + 8],  gb, off + 16);
        async_copy_b128(&sQ[row][ch + 16], gb, off + 32);
        async_copy_b128(&sQ[row][ch + 24], gb, off + 48);
    }
    stageK(0, 0);
    loadV(0);
    async_wait0();
    __syncthreads();

    // Q as WMMA-B operand (lane = query col); scale = (1/sqrt(HD)) * log2(e)
    Frag fqB[2];
    {
        const float QSCALE = 0.125f * 1.44269504088896340736f;
        int hb = (lane >> 4) * 16;
#pragma unroll
        for (int s = 0; s < 2; s++)
#pragma unroll
            for (int e = 0; e < 16; e++)
                fqB[s].h[e] = f2bf(bf2f(sQ[qown][s * 32 + hb + e]) * QSCALE);
    }
    storeV(0);

    float mq = -__builtin_inff();   // per-lane: its query's running max (log2 domain)
    float lq = 0.f;                 // per-lane: its query's running sum
    v8f oaccT[4];                   // O^T: row hd = nt*16 + r + (lane>>4)*8 ; col = query
#pragma unroll
    for (int i = 0; i < 4; i++)
#pragma unroll
        for (int r = 0; r < 8; r++) oaccT[i][r] = 0.f;

    for (int it = 0; it < NIT; it++) {
        const int b = it & 1;
        async_wait0();       // stage(it) K DMA complete (this wave)
        __syncthreads();     // all waves: K(it)/V(it) in LDS, compute(it-1) done
        if (it + 1 < NIT) {  // overlaps compute(it)
            stageK(it + 1, b ^ 1);
            loadV(it + 1);
        }

        // ---- S^T tiles: st[mt] = K(mt-th 16 keys) x Q(16 queries) ----
        v8f st[4];
#pragma unroll
        for (int i = 0; i < 4; i++)
#pragma unroll
            for (int r = 0; r < 8; r++) st[i][r] = 0.f;
#pragma unroll
        for (int mt = 0; mt < 4; mt++)
#pragma unroll
            for (int s = 0; s < 2; s++) {
                Frag fk;   // A operand: K rows (key as WMMA-M)
                int m  = mt * 16 + (lane & 15);
                int kb = s * 32 + (lane >> 4) * 16;
#pragma unroll
                for (int j = 0; j < 4; j++) {
                    fk.u[j]     = *(const unsigned int*)&sK[b][m][kb + 2 * j];
                    fk.u[4 + j] = *(const unsigned int*)&sK[b][m][kb + 8 + 2 * j];
                }
                st[mt] = __builtin_amdgcn_wmma_f32_16x16x32_bf16(
                    false, fk.b, false, fqB[s].b, (short)0, st[mt], false, false);
            }

        // ---- online softmax: within-lane tree + one xor-16 shuffle ----
        float mx = st[0][0];
#pragma unroll
        for (int i = 0; i < 4; i++)
#pragma unroll
            for (int r = 0; r < 8; r++) mx = fmaxf(mx, st[i][r]);
        mx = fmaxf(mx, __shfl_xor(mx, 16, 32));
        float mnew = fmaxf(mq, mx);
        float corr = exp2f(mq - mnew);
        float ps = 0.f;
#pragma unroll
        for (int i = 0; i < 4; i++)
#pragma unroll
            for (int r = 0; r < 8; r++) {
                float p = exp2f(st[i][r] - mnew);
                st[i][r] = p;
                ps += p;
            }
        ps += __shfl_xor(ps, 16, 32);
        lq = lq * corr + ps;
        mq = mnew;
#pragma unroll
        for (int i = 0; i < 4; i++)
#pragma unroll
            for (int r = 0; r < 8; r++) oaccT[i][r] *= corr;

        // ---- P^T (C layout) -> sP[query][key] (per-wave region of sQ) ----
        {
            int kb = (lane >> 4) * 8;
#pragma unroll
            for (int mt = 0; mt < 4; mt++)
#pragma unroll
                for (int r = 0; r < 8; r++)
                    sQ[qown][mt * 16 + kb + r] = f2bf(st[mt][r]);
        }
        // same-wave LDS ops are in-order: reads below observe our writes

        // ---- O^T += V^T(hd x keys) @ P^T(keys x queries) ----
#pragma unroll
        for (int kc = 0; kc < 2; kc++) {
            Frag fp;   // B operand from P^T (keys contiguous per query row)
            {
                int kb2 = kc * 32 + (lane >> 4) * 16;
#pragma unroll
                for (int j = 0; j < 8; j++)
                    fp.u[j] = *(const unsigned int*)&sQ[qown][kb2 + 2 * j];
            }
#pragma unroll
            for (int nt = 0; nt < 4; nt++) {
                Frag fv;   // A operand from V^T (hd as WMMA-M)
                int hd = nt * 16 + (lane & 15);
                int kb = kc * 32 + (lane >> 4) * 16;
#pragma unroll
                for (int j = 0; j < 4; j++) {
                    fv.u[j]     = *(const unsigned int*)&sVt[b][hd][kb + 2 * j];
                    fv.u[4 + j] = *(const unsigned int*)&sVt[b][hd][kb + 8 + 2 * j];
                }
                oaccT[nt] = __builtin_amdgcn_wmma_f32_16x16x32_bf16(
                    false, fv.b, false, fp.b, (short)0, oaccT[nt], false, false);
            }
        }

        if (it + 1 < NIT) storeV(b ^ 1);   // loadcnt wait hidden under compute
    }

    {   // normalize + store: lane owns one query row -> packed b128 stores
        float inv = 1.f / lq;
        size_t base = (rowbase + (size_t)qblk * BQ + qown) * DD + colbase + (lane >> 4) * 8;
#pragma unroll
        for (int nt = 0; nt < 4; nt++) {
            union { uint4 v; unsigned short h[8]; } pk;
#pragma unroll
            for (int r = 0; r < 8; r++) pk.h[r] = f2bf(oaccT[nt][r] * inv);
            *(uint4*)(Ob + base + nt * 16) = pk.v;
        }
    }
}

// ---------------------------------------------------------------- launch ----
extern "C" void kernel_launch(void* const* d_in, const int* in_sizes, int n_in,
                              void* d_out, int out_size, void* d_ws, size_t ws_size,
                              hipStream_t stream) {
    const float* Q  = (const float*)d_in[0];
    const float* Wk = (const float*)d_in[1];
    const float* Wv = (const float*)d_in[2];
    const float* Wo = (const float*)d_in[3];
    float* out = (float*)d_out;

    // workspace layout (bf16 halfwords), ~33.5 MiB total
    unsigned short* Qb  = (unsigned short*)d_ws;
    unsigned short* Kb  = Qb  + (size_t)MROWS * DD;
    unsigned short* Vb  = Kb  + (size_t)MROWS * DD;
    unsigned short* Ab  = Vb  + (size_t)MROWS * DD;
    unsigned short* Wkb = Ab  + (size_t)MROWS * DD;
    unsigned short* Wvb = Wkb + (size_t)DD * DD;
    unsigned short* Wob = Wvb + (size_t)DD * DD;

    const int nQ = MROWS * DD;
    const int nW = DD * DD;
    cast_f32_bf16_kernel<<<(nQ + 255) / 256, 256, 0, stream>>>(Q, Qb, nQ);
    cast_f32_bf16_kernel<<<(nW + 255) / 256, 256, 0, stream>>>(Wk, Wkb, nW);
    cast_f32_bf16_kernel<<<(nW + 255) / 256, 256, 0, stream>>>(Wv, Wvb, nW);
    cast_f32_bf16_kernel<<<(nW + 255) / 256, 256, 0, stream>>>(Wo, Wob, nW);

    dim3 ggrid(DD / GN, MROWS / GM);   // (8, 64)
    // keys = Q @ Wk^T + Q (fused residual), values = Q @ Wv^T
    gemm_bf16_kernel<1><<<ggrid, 256, 0, stream>>>(Qb, Wkb, Qb, Kb, nullptr);
    gemm_bf16_kernel<0><<<ggrid, 256, 0, stream>>>(Qb, Wvb, nullptr, Vb, nullptr);

    dim3 agrid(TT / BQ, NB, HH);       // (16, 4, 8)
    flash_attn_kernel<<<agrid, 256, 0, stream>>>(Qb, Kb, Vb, Ab);

    // out = attn @ Wo^T, fp32 store
    gemm_bf16_kernel<2><<<ggrid, 256, 0, stream>>>(Ab, Wob, nullptr, nullptr, out);
}
